// Gemma3Attention_89996744721049
// MI455X (gfx1250) — matrix operand order repaired and checked by reading, not compiled, not run on hardware
//
#include <hip/hip_runtime.h>

// ---------------------------------------------------------------------------
// Gemma3 sliding-window attention layer for gfx1250 (MI455X).
// bf16 WMMA (v_wmma_f32_16x16x32_bf16) for all GEMMs, f32 accumulate.
// Async global->LDS DMA (ASYNCcnt) with double-buffered LDS in the GEMM.
// ---------------------------------------------------------------------------

typedef __bf16 bf16;
typedef __bf16  v16bf __attribute__((ext_vector_type(16)));
typedef float   v8f   __attribute__((ext_vector_type(8)));

#define NUM_HEADS    16
#define NUM_KV_HEADS 8
#define HEAD_DIM     128
#define WINDOW       1024
#define SEQ          2048
#define BATCH        2
#define HID          2048
#define MTOT         (BATCH * SEQ)             // 4096 rows
#define QDIM         (NUM_HEADS * HEAD_DIM)    // 2048
#define KVDIM        (NUM_KV_HEADS * HEAD_DIM) // 1024

union FragBF {
    v16bf v;
    bf16  h[16];
    uint4 q[2];
};

// A-fragment (16x32, M x K), per CDNA5 ISA 7.12.2:
//  lanes 0-15:  M=lane,   K chunks [0..7]  and [16..23]
//  lanes 16-31: M=lane-16,K chunks [8..15] and [24..31]
__device__ __forceinline__ FragBF load_a_frag(const bf16* base, int stride, int kbase) {
    const int lane = threadIdx.x & 31;
    const int hf = lane >> 4, ln = lane & 15;
    FragBF f;
    const bf16* rp = base + (size_t)ln * stride + kbase + 8 * hf;
    f.q[0] = *(const uint4*)rp;
    f.q[1] = *(const uint4*)(rp + 16);
    return f;
}

// B-fragment (32x16, K x N), element B[k][n] stored at base[n*stride + k]:
//  lane L: N = L%16, K chunks [8h..8h+7] and [16+8h..16+8h+7], h = L/16
__device__ __forceinline__ FragBF load_b_frag(const bf16* base, int stride, int kbase) {
    const int lane = threadIdx.x & 31;
    const int hf = lane >> 4, ln = lane & 15;
    FragBF f;
    const bf16* rp = base + (size_t)ln * stride + kbase + 8 * hf;
    f.q[0] = *(const uint4*)rp;
    f.q[1] = *(const uint4*)(rp + 16);
    return f;
}

__device__ __forceinline__ v8f wmma_bf16(const FragBF& a, const FragBF& b, v8f c) {
    return __builtin_amdgcn_wmma_f32_16x16x32_bf16(
        /*neg_a=*/false, a.v, /*neg_b=*/false, b.v,
        /*c_mod=*/(short)0, c, /*reuse_a=*/false, /*reuse_b=*/false);
}

// ---- CDNA5 async global->LDS DMA (tracked by ASYNCcnt) --------------------
// LDS offset = low 32 bits of the generic shared pointer (wave-relative).
__device__ __forceinline__ void async_load_b128(const bf16* lds_dst, const bf16* gsrc) {
    unsigned int  loff = (unsigned int)(uintptr_t)lds_dst;
    unsigned long long ga = (unsigned long long)(uintptr_t)gsrc;
    asm volatile("global_load_async_to_lds_b128 %0, %1, off"
                 :: "v"(loff), "v"(ga) : "memory");
}
__device__ __forceinline__ void wait_async_le4() {   // current slice done (in-order)
    asm volatile("s_wait_asynccnt 0x4" ::: "memory");
}
__device__ __forceinline__ void wait_async_0() {
    asm volatile("s_wait_asynccnt 0x0" ::: "memory");
}

// ---------------------------------------------------------------------------
// f32 -> bf16 conversion (vectorized x4)
// ---------------------------------------------------------------------------
struct B4 { bf16 a, b, c, d; };

__global__ void cvt4_f32_bf16(const float4* __restrict__ in, B4* __restrict__ out, int n4) {
    int i = blockIdx.x * 256 + threadIdx.x;
    if (i < n4) {
        float4 v = in[i];
        B4 o = { (bf16)v.x, (bf16)v.y, (bf16)v.z, (bf16)v.w };
        out[i] = o;
    }
}

// ---------------------------------------------------------------------------
// GEMM: C[m,n] = sum_k A[m,k] * W[n,k]   (A: MxK bf16 row-major, W: NxK bf16)
// Block tile 128x128, 8 waves. Double-buffered LDS, async DMA staging.
// mode 0: store bf16           (V projection)
// mode 1: RMSNorm+RoPE -> bf16 (Q/K projections; block's 128 cols == 1 head)
// mode 2: store f32            (output projection -> d_out)
// ---------------------------------------------------------------------------
__global__ __launch_bounds__(256) void gemm_bf16_kernel(
    const bf16* __restrict__ A, const bf16* __restrict__ W,
    bf16* __restrict__ outB, float* __restrict__ outF,
    const float* __restrict__ normw,
    int K, int ostride, int mode, int seqMask)
{
    __shared__ bf16 As[2][128 * 32];
    __shared__ bf16 Ws[2][128 * 32];

    const int tid = threadIdx.x;
    const int wave = tid >> 5;
    const int lane = tid & 31;
    const int hf = lane >> 4, ln = lane & 15;
    const int m0 = blockIdx.x * 128;
    const int n0 = blockIdx.y * 128;

    v8f acc[8];
#pragma unroll
    for (int j = 0; j < 8; ++j) acc[j] = (v8f){0,0,0,0,0,0,0,0};

    // Per-thread staging coordinates: 2 b128 chunks per array per slice.
    const int r0 = tid >> 2,          c0 = (tid & 3) * 8;
    const int r1 = (tid + 256) >> 2,  c1 = ((tid + 256) & 3) * 8;

    // Prologue: DMA slice 0 into buffer 0 (4 async loads per thread).
    async_load_b128(&As[0][r0 * 32 + c0], &A[(size_t)(m0 + r0) * K + c0]);
    async_load_b128(&Ws[0][r0 * 32 + c0], &W[(size_t)(n0 + r0) * K + c0]);
    async_load_b128(&As[0][r1 * 32 + c1], &A[(size_t)(m0 + r1) * K + c1]);
    async_load_b128(&Ws[0][r1 * 32 + c1], &W[(size_t)(n0 + r1) * K + c1]);

    const int nk = K >> 5;
    int buf = 0;
    for (int t = 0; t < nk; ++t) {
        __syncthreads();   // all waves finished reading the buffer we now overwrite
        if (t + 1 < nk) {
            const int kb = (t + 1) * 32;
            async_load_b128(&As[buf ^ 1][r0 * 32 + c0], &A[(size_t)(m0 + r0) * K + kb + c0]);
            async_load_b128(&Ws[buf ^ 1][r0 * 32 + c0], &W[(size_t)(n0 + r0) * K + kb + c0]);
            async_load_b128(&As[buf ^ 1][r1 * 32 + c1], &A[(size_t)(m0 + r1) * K + kb + c1]);
            async_load_b128(&Ws[buf ^ 1][r1 * 32 + c1], &W[(size_t)(n0 + r1) * K + kb + c1]);
            wait_async_le4();   // retire current slice; next slice stays in flight
        } else {
            wait_async_0();
        }
        __syncthreads();        // current slice visible to all waves

        FragBF af = load_a_frag(&As[buf][(16 * wave) * 32], 32, 0);
        FragBF bfr[8];
#pragma unroll
        for (int j = 0; j < 8; ++j)
            bfr[j] = load_b_frag(&Ws[buf][(16 * j) * 32], 32, 0);
#pragma unroll
        for (int j = 0; j < 8; ++j)
            acc[j] = wmma_bf16(af, bfr[j], acc[j]);
        buf ^= 1;
    }

    const int mbase = m0 + 16 * wave + 8 * hf;   // global row of acc[.][r] for this lane

    if (mode == 1) {
        // Gemma RMSNorm over head_dim (=128 block cols), then RoPE (theta=1e4).
#pragma unroll
        for (int r = 0; r < 8; ++r) {
            float ss = 0.f;
#pragma unroll
            for (int j = 0; j < 8; ++j) ss += acc[j][r] * acc[j][r];
            ss += __shfl_xor(ss, 1);
            ss += __shfl_xor(ss, 2);
            ss += __shfl_xor(ss, 4);
            ss += __shfl_xor(ss, 8);          // reduce within 16-lane half (same row)
            float scale = __frsqrt_rn(ss * (1.0f / 128.0f) + 1e-6f);
#pragma unroll
            for (int j = 0; j < 8; ++j) {
                float w = normw[16 * j + ln];
                acc[j][r] = acc[j][r] * scale * (1.0f + w);
            }
            // RoPE: pair (c, c+64) => tiles (j, j+4); cos/sin share freq index c
            int pos = (mbase + r) & seqMask;
            float fpos = (float)pos;
#pragma unroll
            for (int j = 0; j < 4; ++j) {
                int c = 16 * j + ln;                         // 0..63
                float invf = __expf(-(float)c * 0.14391156831212787f); // ln(1e4)/64
                float s, co;
                __sincosf(fpos * invf, &s, &co);
                float x1 = acc[j][r], x2 = acc[j + 4][r];
                acc[j][r]     = x1 * co - x2 * s;
                acc[j + 4][r] = x2 * co + x1 * s;
            }
        }
    }

#pragma unroll
    for (int r = 0; r < 8; ++r) {
        int row = mbase + r;
#pragma unroll
        for (int j = 0; j < 8; ++j) {
            int col = n0 + 16 * j + ln;
            if (mode == 2) outF[(size_t)row * ostride + col] = acc[j][r];
            else           outB[(size_t)row * ostride + col] = (bf16)acc[j][r];
        }
    }
}

// ---------------------------------------------------------------------------
// Flash-style sliding-window attention. Grid: (S/128, heads, batch).
// 8 waves; wave w owns 16 q-rows. K tile DMA'd async into LDS (overlapping the
// manual transposed-V staging); P scores round-trip through LDS to convert
// C-layout -> A-fragment layout (per-wave DS ops are in-order).
// ---------------------------------------------------------------------------
#define VSTR 80   // padded Vt row stride (halfs); 160B rows keep b128 alignment

__global__ __launch_bounds__(256) void attn_kernel(
    const bf16* __restrict__ Q, const bf16* __restrict__ Km,
    const bf16* __restrict__ Vm, bf16* __restrict__ O)
{
    __shared__ bf16 Ks[64 * 128];        // [key][d]
    __shared__ bf16 Vt[128 * VSTR];      // [d][key]
    __shared__ bf16 Ps[8 * 16 * 64];     // per-wave P tile [16 q][64 key]

    const int tid = threadIdx.x;
    const int wave = tid >> 5;
    const int lane = tid & 31;
    const int hf = lane >> 4, ln = lane & 15;

    const int qs   = blockIdx.x * 128;
    const int head = blockIdx.y;
    const int b    = blockIdx.z;
    const int kvh  = head >> 1;                    // repeat_interleave(groups=2)

    // Q A-fragments: rows qs+16w.., cols head*128 + 32k..  (loaded once)
    const bf16* qbase = Q + ((size_t)(b * SEQ + qs + 16 * wave)) * QDIM + head * HEAD_DIM;
    FragBF qa[4];
#pragma unroll
    for (int ks = 0; ks < 4; ++ks) qa[ks] = load_a_frag(qbase, QDIM, 32 * ks);

    float rm[8], rl[8];
    v8f o[8];
#pragma unroll
    for (int r = 0; r < 8; ++r) { rm[r] = -1e30f; rl[r] = 0.f; }
#pragma unroll
    for (int dt = 0; dt < 8; ++dt) o[dt] = (v8f){0,0,0,0,0,0,0,0};

    const int kb0 = (qs > (WINDOW - 1)) ? ((qs - (WINDOW - 1)) & ~63) : 0;
    const float sc = 0.08838834764831845f;  // 1/sqrt(128)

    for (int kb = kb0; kb < qs + 128; kb += 64) {
        __syncthreads();
        // Async DMA the K tile (64 keys x 128 d) straight into LDS...
#pragma unroll
        for (int i = 0; i < 4; ++i) {
            int e = tid + 256 * i;               // 0..1023 b128 chunks
            int r = e >> 4, c = (e & 15) * 8;    // key row 0..63, d col
            size_t grow = (size_t)(b * SEQ + kb + r) * KVDIM + kvh * HEAD_DIM + c;
            async_load_b128(&Ks[r * 128 + c], &Km[grow]);
        }
        // ...while we stage V transposed through VGPRs (scatter stores).
#pragma unroll
        for (int i = 0; i < 4; ++i) {
            int e = tid + 256 * i;
            int r = e >> 4, c = (e & 15) * 8;
            size_t grow = (size_t)(b * SEQ + kb + r) * KVDIM + kvh * HEAD_DIM + c;
            uint4 vx = *(const uint4*)&Vm[grow];
            const bf16* vh = (const bf16*)&vx;
#pragma unroll
            for (int q = 0; q < 8; ++q) Vt[(c + q) * VSTR + r] = vh[q];
        }
        wait_async_0();
        __syncthreads();

        // Scores: S[16q x 64k] = Q * K^T  (contraction over d=128)
        v8f st[4];
#pragma unroll
        for (int nt = 0; nt < 4; ++nt) {
            st[nt] = (v8f){0,0,0,0,0,0,0,0};
#pragma unroll
            for (int ks = 0; ks < 4; ++ks) {
                FragBF kf = load_b_frag(&Ks[(16 * nt) * 128], 128, 32 * ks);
                st[nt] = wmma_bf16(qa[ks], kf, st[nt]);
            }
        }

        // Mask + online softmax + P to LDS (bf16)
#pragma unroll
        for (int r = 0; r < 8; ++r) {
            int row = qs + 16 * wave + 8 * hf + r;
#pragma unroll
            for (int nt = 0; nt < 4; ++nt) {
                int col = kb + 16 * nt + ln;
                bool ok = (col <= row) && (col > row - WINDOW);
                st[nt][r] = ok ? st[nt][r] * sc : -1e30f;
            }
            float pm = fmaxf(fmaxf(st[0][r], st[1][r]), fmaxf(st[2][r], st[3][r]));
            pm = fmaxf(pm, __shfl_xor(pm, 1));
            pm = fmaxf(pm, __shfl_xor(pm, 2));
            pm = fmaxf(pm, __shfl_xor(pm, 4));
            pm = fmaxf(pm, __shfl_xor(pm, 8));
            float rmn = fmaxf(rm[r], pm);
            float alpha = __expf(rm[r] - rmn);
            rm[r] = rmn;
            float psum = 0.f;
#pragma unroll
            for (int nt = 0; nt < 4; ++nt) {
                float p = (st[nt][r] <= -0.5e30f) ? 0.f : __expf(st[nt][r] - rmn);
                st[nt][r] = p;
                psum += p;
            }
            psum += __shfl_xor(psum, 1);
            psum += __shfl_xor(psum, 2);
            psum += __shfl_xor(psum, 4);
            psum += __shfl_xor(psum, 8);
            rl[r] = rl[r] * alpha + psum;
#pragma unroll
            for (int dt = 0; dt < 8; ++dt) o[dt][r] *= alpha;
#pragma unroll
            for (int nt = 0; nt < 4; ++nt)
                Ps[wave * 1024 + (8 * hf + r) * 64 + 16 * nt + ln] = (bf16)st[nt][r];
        }

        // O += P * V  (contraction over 64 keys; per-wave DS ops are in-order)
        FragBF pf0 = load_a_frag(&Ps[wave * 1024], 64, 0);
        FragBF pf1 = load_a_frag(&Ps[wave * 1024], 64, 32);
#pragma unroll
        for (int dt = 0; dt < 8; ++dt) {
            FragBF v0 = load_b_frag(&Vt[(16 * dt) * VSTR], VSTR, 0);
            o[dt] = wmma_bf16(pf0, v0, o[dt]);
            FragBF v1 = load_b_frag(&Vt[(16 * dt) * VSTR], VSTR, 32);
            o[dt] = wmma_bf16(pf1, v1, o[dt]);
        }
    }

    // Epilogue: O / rowsum -> bf16 attn buffer [token][head*128 + d]
#pragma unroll
    for (int r = 0; r < 8; ++r) {
        float inv = 1.0f / rl[r];
        int row = b * SEQ + qs + 16 * wave + 8 * hf + r;
#pragma unroll
        for (int dt = 0; dt < 8; ++dt) {
            int col = head * HEAD_DIM + 16 * dt + ln;
            O[(size_t)row * QDIM + col] = (bf16)(o[dt][r] * inv);
        }
    }
}

// ---------------------------------------------------------------------------
// Host launcher
// ---------------------------------------------------------------------------
extern "C" void kernel_launch(void* const* d_in, const int* in_sizes, int n_in,
                              void* d_out, int out_size, void* d_ws, size_t ws_size,
                              hipStream_t stream) {
    (void)in_sizes; (void)n_in; (void)out_size; (void)ws_size;

    const float* hs = (const float*)d_in[0];   // [2,2048,2048]
    const float* wq = (const float*)d_in[1];   // [2048,2048]
    const float* wk = (const float*)d_in[2];   // [1024,2048]
    const float* wv = (const float*)d_in[3];   // [1024,2048]
    const float* wo = (const float*)d_in[4];   // [2048,2048]
    const float* qn = (const float*)d_in[5];   // [128]
    const float* kn = (const float*)d_in[6];   // [128]
    float* out = (float*)d_out;                // [2,2048,2048] f32

    // Workspace layout (bf16 elements)
    bf16* ws   = (bf16*)d_ws;
    bf16* xb   = ws;                                   // 4096x2048
    bf16* wqb  = xb  + (size_t)MTOT * HID;             // 2048x2048
    bf16* wkb  = wqb + (size_t)QDIM * HID;             // 1024x2048
    bf16* wvb  = wkb + (size_t)KVDIM * HID;            // 1024x2048
    bf16* wob  = wvb + (size_t)KVDIM * HID;            // 2048x2048
    bf16* qb   = wob + (size_t)HID * QDIM;             // 4096x2048
    bf16* kbf  = qb  + (size_t)MTOT * QDIM;            // 4096x1024
    bf16* vbf  = kbf + (size_t)MTOT * KVDIM;           // 4096x1024
    bf16* ab   = vbf + (size_t)MTOT * KVDIM;           // 4096x2048

    // 1) f32 -> bf16 conversions
    auto cvt = [&](const float* src, bf16* dst, size_t n) {
        int n4 = (int)(n / 4);
        cvt4_f32_bf16<<<(n4 + 255) / 256, 256, 0, stream>>>(
            (const float4*)src, (B4*)dst, n4);
    };
    cvt(hs, xb,  (size_t)MTOT * HID);
    cvt(wq, wqb, (size_t)QDIM * HID);
    cvt(wk, wkb, (size_t)KVDIM * HID);
    cvt(wv, wvb, (size_t)KVDIM * HID);
    cvt(wo, wob, (size_t)HID * QDIM);

    // 2) Q projection + RMSNorm + RoPE (each grid.y block = one head)
    gemm_bf16_kernel<<<dim3(MTOT / 128, NUM_HEADS), 256, 0, stream>>>(
        xb, wqb, qb, nullptr, qn, HID, QDIM, /*mode=*/1, SEQ - 1);
    // 3) K projection + RMSNorm + RoPE
    gemm_bf16_kernel<<<dim3(MTOT / 128, NUM_KV_HEADS), 256, 0, stream>>>(
        xb, wkb, kbf, nullptr, kn, HID, KVDIM, /*mode=*/1, SEQ - 1);
    // 4) V projection (plain)
    gemm_bf16_kernel<<<dim3(MTOT / 128, NUM_KV_HEADS), 256, 0, stream>>>(
        xb, wvb, vbf, nullptr, nullptr, HID, KVDIM, /*mode=*/0, SEQ - 1);

    // 5) Sliding-window attention
    attn_kernel<<<dim3(SEQ / 128, NUM_HEADS, BATCH), 256, 0, stream>>>(
        qb, kbf, vbf, ab);

    // 6) Output projection -> f32 d_out
    gemm_bf16_kernel<<<dim3(MTOT / 128, HID / 128), 256, 0, stream>>>(
        ab, wob, nullptr, out, nullptr, QDIM, HID, /*mode=*/2, SEQ - 1);
}
